// ConvLSTMencode_15874199126027
// MI455X (gfx1250) — compile-verified, matrix operand order checked by reading, not run on hardware
//
#include <hip/hip_runtime.h>

typedef __attribute__((ext_vector_type(16))) _Float16 v16h;
typedef __attribute__((ext_vector_type(8)))  float    v8f;
typedef __attribute__((ext_vector_type(4)))  unsigned int v4u;

union FragU { v4u u[2]; v16h h; };

__device__ __forceinline__ float hsig(float x) {
    return fminf(fmaxf(0.2f * x + 0.5f, 0.f), 1.f);
}

// ---- gfx1250 async global->LDS helpers -------------------------------------
__device__ __forceinline__ void async_load_b128(const void* gptr, unsigned lds_off) {
    // GLOBAL_LOAD_ASYNC_TO_LDS_B128: vDst = LDS byte address, vAddr = 64-bit global addr
    asm volatile("global_load_async_to_lds_b128 %0, %1, off"
                 :: "v"(lds_off), "v"(gptr) : "memory");
}
__device__ __forceinline__ void wait_async_le(int) {}
__device__ __forceinline__ void wait_async_2() {
#if defined(__has_builtin) && __has_builtin(__builtin_amdgcn_s_wait_asynccnt)
    __builtin_amdgcn_s_wait_asynccnt(2);
#else
    asm volatile("s_wait_asynccnt 0x2" ::: "memory");
#endif
}
__device__ __forceinline__ void wait_async_0() {
#if defined(__has_builtin) && __has_builtin(__builtin_amdgcn_s_wait_asynccnt)
    __builtin_amdgcn_s_wait_asynccnt(0);
#else
    asm volatile("s_wait_asynccnt 0x0" ::: "memory");
#endif
}
__device__ __forceinline__ unsigned lds_offset_of(const void* p) {
    // addrspacecast(AS3->generic) keeps the LDS byte offset in the low 32 bits
    return (unsigned)(uintptr_t)p;
}

// ---------------------------------------------------------------------------
// Pack (W;U) f32 HWIO weights into per-block-contiguous WMMA B-fragment order.
// Layout: dst[(((kt*(F/16) + nty)*4 + g) * 512) + lane*16 + i] where
//   k = kt*32 + basek(lane) + (i<8 ? i : i+8),  n = g*F + nty*16 + (lane&15).
// One K-step's B data for one block (4 gates) is a contiguous 4 KB chunk.
// ---------------------------------------------------------------------------
__global__ void pack_weights(const float* __restrict__ Wsrc,
                             const float* __restrict__ Usrc,
                             _Float16* __restrict__ dst,
                             int Cin, int F, int Kpad)
{
    int tid = blockIdx.x * blockDim.x + threadIdx.x;
    int total = Kpad * 4 * F;
    if (tid >= total) return;
    int i    = tid & 15;
    int lane = (tid >> 4) & 31;
    int frag = tid >> 9;
    int g    = frag & 3;
    int q    = frag >> 2;
    int nTy  = F >> 4;
    int nty  = q % nTy;
    int kt   = q / nTy;
    int basek = (lane < 16) ? 0 : 8;
    int k = kt * 32 + basek + ((i < 8) ? i : i + 8);
    int n = g * F + nty * 16 + (lane & 15);
    int Kx = 9 * Cin, Ktot = 9 * (Cin + F);
    float v = 0.f;
    if (k < Kx)        v = Wsrc[(size_t)k * (4 * F) + n];
    else if (k < Ktot) v = Usrc[(size_t)(k - Kx) * (4 * F) + n];
    dst[tid] = (_Float16)v;
}

__global__ void f32_to_f16_kernel(const float* __restrict__ in,
                                  _Float16* __restrict__ out, int n)
{
    int i = blockIdx.x * blockDim.x + threadIdx.x;
    if (i < n) out[i] = (_Float16)in[i];
}

__global__ void zero_state(_Float16* __restrict__ h, float* __restrict__ c, int n)
{
    int i = blockIdx.x * blockDim.x + threadIdx.x;
    if (i < n) { h[i] = (_Float16)0.f; c[i] = 0.f; }
}

// ---------------------------------------------------------------------------
// Fused ConvLSTM cell step (templated on the compile-time layer config).
// Implicit-GEMM over K = 9*(Cin+F) padded to Kpad; 4 WMMA column tiles per
// wave (one per gate); B fragments staged via double-buffered async
// global->LDS loads; fused gate epilogue.
// Block: 128 threads = 4 waves. Tile: 64 pixels x 16 gate channels.
// Grid: (H*W/64, F/16, B)
// ---------------------------------------------------------------------------
template<int H, int W, int Cin, int F, int Kpad>
__global__ __launch_bounds__(128)
void convlstm_cell(const _Float16* __restrict__ xin,   // (B,T,H,W,Cin) f16
                   const _Float16* __restrict__ hin,   // (B,H,W,F) f16
                   _Float16* __restrict__ hout,        // (B,H,W,F) f16
                   float* __restrict__ cst,            // (B,H,W,F) f32
                   const _Float16* __restrict__ wpack, // packed B fragments
                   const float* __restrict__ bias,     // (4F)
                   _Float16* __restrict__ out,         // (B,T,H,W,F) f16
                   int t_in, int s_out)
{
    constexpr int T    = 10;
    constexpr int HW   = H * W;
    constexpr int Kx   = 9 * Cin;
    constexpr int Ktot = Kx + 9 * F;
    constexpr int nK   = Kpad / 32;

    __shared__ __align__(16) _Float16 Atile[64 * 32];        // 4 KB
    __shared__ __align__(16) _Float16 Btile[2][4 * 512];     // 2 x 4 KB

    const int m0   = blockIdx.x * 64;
    const int by   = blockIdx.y;
    const int b    = blockIdx.z;
    const int tid  = threadIdx.x;
    const int wave = tid >> 5;
    const int lane = tid & 31;

    const v8f zf8 = {0.f, 0.f, 0.f, 0.f, 0.f, 0.f, 0.f, 0.f};
    v8f acc[4];
    acc[0] = zf8; acc[1] = zf8; acc[2] = zf8; acc[3] = zf8;

    const _Float16* xframe = xin + (size_t)(b * T + t_in) * HW * Cin;
    const _Float16* hframe = hin + (size_t)b * HW * F;

    // per-thread im2col fill coordinates (2 threads per pixel row of tile)
    const int ml   = tid >> 1;
    const int kk0  = (tid & 1) << 4;
    const int pix0 = m0 + ml;
    const int py0  = pix0 / W;              // W is power of two -> shift
    const int px0  = pix0 & (W - 1);

    // per-thread async copy slots for the block's 4 KB B chunk
    const unsigned ldsB0 = lds_offset_of(&Btile[0][0]);
    const unsigned ldsB1 = lds_offset_of(&Btile[1][0]);
    const char* wbase = (const char*)wpack + (size_t)by * 4096;  // chunk for nty=by, kt=0
    constexpr size_t ktStride = (size_t)(F >> 4) * 4096;          // bytes per kt step

    // prologue: kick off B chunk for kt = 0 into buffer 0
    async_load_b128(wbase + (size_t)tid * 16,        ldsB0 + tid * 16);
    async_load_b128(wbase + 2048 + (size_t)tid * 16, ldsB0 + 2048 + tid * 16);

    for (int kt = 0; kt < nK; ++kt) {
        // ---- cooperative im2col fill of 64x32 f16 A tile ----
        const int kb = kt * 32;
        #pragma unroll
        for (int u = 0; u < 16; ++u) {
            int k = kb + kk0 + u;
            _Float16 v = (_Float16)0.f;
            if (k < Ktot) {
                int tap, c;
                bool isX = (k < Kx);
                if (isX) { tap = k / Cin; c = k - tap * Cin; }      // constant divisors
                else     { int kh = k - Kx; tap = kh / F; c = kh - tap * F; }
                int y = py0 + tap / 3 - 1;
                int x = px0 + tap % 3 - 1;
                if ((unsigned)y < (unsigned)H && (unsigned)x < (unsigned)W) {
                    if (isX) v = xframe[(size_t)(y * W + x) * Cin + c];
                    else     v = hframe[(size_t)(y * W + x) * F + c];
                }
            }
            Atile[ml * 32 + kk0 + u] = v;
        }

        // ---- prefetch next B chunk (async), then wait for current ----
        if (kt + 1 < nK) {
            const char* g = wbase + (size_t)(kt + 1) * ktStride;
            unsigned l = ((kt + 1) & 1) ? ldsB1 : ldsB0;
            async_load_b128(g + (size_t)tid * 16,        l + tid * 16);
            async_load_b128(g + 2048 + (size_t)tid * 16, l + 2048 + tid * 16);
            wait_async_2();          // previous 2 (current chunk) are complete
        } else {
            wait_async_0();
        }
        __syncthreads();             // A tile stores + all waves' async B ready

        // ---- A fragment: two ds_load_b128 per lane ----
        FragU af;
        {
            int mrow  = wave * 16 + (lane & 15);
            int basek = (lane < 16) ? 0 : 8;
            af.u[0] = *(const v4u*)&Atile[mrow * 32 + basek];
            af.u[1] = *(const v4u*)&Atile[mrow * 32 + basek + 16];
        }

        // ---- B fragments from LDS + 4 WMMAs (one per gate) ----
        const _Float16* bcur = &Btile[kt & 1][0];
        #pragma unroll
        for (int g = 0; g < 4; ++g) {
            FragU bf;
            const v4u* bp = (const v4u*)&bcur[g * 512 + (lane << 4)];
            bf.u[0] = bp[0];
            bf.u[1] = bp[1];
            acc[g] = __builtin_amdgcn_wmma_f32_16x16x32_f16(
                false, af.h, false, bf.h, (short)0, acc[g], false, false);
        }
        __syncthreads();             // before next iteration's A fill
    }

    // ---- fused gate epilogue ----
    const int j = by * 16 + (lane & 15);
    const float bi  = bias[0 * F + j];
    const float bfv = bias[1 * F + j];
    const float bg  = bias[2 * F + j];
    const float bo  = bias[3 * F + j];

    const int mbase = m0 + wave * 16 + ((lane < 16) ? 0 : 8);
    _Float16* outframe = out  + (size_t)(b * T + s_out) * HW * F;
    float*    cframe   = cst  + (size_t)b * HW * F;
    _Float16* hfw      = hout + (size_t)b * HW * F;

    #pragma unroll
    for (int r = 0; r < 8; ++r) {
        int pix = mbase + r;
        size_t idx = (size_t)pix * F + j;
        float zi  = acc[0][r] + bi;
        float zfv = acc[1][r] + bfv;
        float zg  = acc[2][r] + bg;
        float zo  = acc[3][r] + bo;
        float ig = hsig(zi), fg = hsig(zfv), og = hsig(zo);
        float gg = fmaxf(zg, 0.f);
        float cold = cframe[idx];
        float cnew = fg * cold + ig * gg;
        float hval = og * fmaxf(cnew, 0.f);
        cframe[idx] = cnew;
        _Float16 h16 = (_Float16)hval;
        hfw[idx] = h16;
        outframe[idx] = h16;
    }
}

// ---------------------------------------------------------------------------
// 2x2 maxpool over spatial dims, NTHWC. Emits f16 (inter-layer) or f32 (final).
// ---------------------------------------------------------------------------
__global__ void maxpool2_kernel(const _Float16* __restrict__ in,
                                _Float16* __restrict__ out16,
                                float* __restrict__ out32,
                                int BT, int H, int W, int C)
{
    int idx = blockIdx.x * blockDim.x + threadIdx.x;
    int Ho = H >> 1, Wo = W >> 1;
    int total = BT * Ho * Wo * C;
    if (idx >= total) return;
    int c  = idx % C;  int t1 = idx / C;
    int xo = t1 % Wo;  int t2 = t1 / Wo;
    int yo = t2 % Ho;  int bt = t2 / Ho;
    const _Float16* base = in + (size_t)bt * H * W * C;
    float m = -3.0e38f;
    #pragma unroll
    for (int dy = 0; dy < 2; ++dy)
        #pragma unroll
        for (int dx = 0; dx < 2; ++dx) {
            float v = (float)base[(size_t)((2 * yo + dy) * W + (2 * xo + dx)) * C + c];
            m = fmaxf(m, v);
        }
    if (out16) out16[idx] = (_Float16)m;
    else       out32[idx] = m;
}

// ---------------------------------------------------------------------------
// Host orchestration
// ---------------------------------------------------------------------------
template<int H, int W, int Cin, int F, int Kpad>
static void run_layer(const _Float16* in, _Float16* outbuf,
                      _Float16* h0, _Float16* h1, float* cst,
                      const _Float16* wp, const float* bias, hipStream_t stream)
{
    constexpr int HW = H * W;
    const int nstate = 4 * HW * F;
    zero_state<<<(nstate + 255) / 256, 256, 0, stream>>>(h0, cst, nstate);
    dim3 grid(HW / 64, F / 16, 4), block(128);
    for (int s = 0; s < 10; ++s) {
        _Float16* hi = (s & 1) ? h1 : h0;
        _Float16* ho = (s & 1) ? h0 : h1;
        convlstm_cell<H, W, Cin, F, Kpad><<<grid, block, 0, stream>>>(
            in, hi, ho, cst, wp, bias, outbuf, /*t_in=*/9 - s, /*s_out=*/s);
    }
}

extern "C" void kernel_launch(void* const* d_in, const int* in_sizes, int n_in,
                              void* d_out, int out_size, void* d_ws, size_t ws_size,
                              hipStream_t stream)
{
    (void)in_sizes; (void)n_in; (void)out_size; (void)ws_size;

    const float* x = (const float*)d_in[0];
    const float* Ws[6]; const float* Us[6]; const float* bs[6];
    for (int i = 0; i < 6; ++i) {
        Ws[i] = (const float*)d_in[1 + 3 * i];
        Us[i] = (const float*)d_in[2 + 3 * i];
        bs[i] = (const float*)d_in[3 + 3 * i];
    }

    // workspace layout (bytes)
    const size_t OFS_B  = 41943040;   // f16 pong buffer (ping at 0)
    const size_t OFS_H0 = 83886080;
    const size_t OFS_H1 = 88080384;
    const size_t OFS_C  = 92274688;
    const size_t OFS_W  = 100663296;

    char* ws = (char*)d_ws;
    _Float16* bufA = (_Float16*)(ws);
    _Float16* bufB = (_Float16*)(ws + OFS_B);
    _Float16* h0   = (_Float16*)(ws + OFS_H0);
    _Float16* h1   = (_Float16*)(ws + OFS_H1);
    float*    cst  = (float*)   (ws + OFS_C);
    _Float16* wp   = (_Float16*)(ws + OFS_W);

    // per-layer packed-weight offsets (halves): Kpad * 4F each
    const size_t wofs[6] = {0, 40960, 114688, 256000, 421888, 684032};
    const int   cins[6]  = {3, 32, 32, 48, 48, 64};
    const int   fs[6]    = {32, 32, 48, 48, 64, 64};
    const int   kpads[6] = {320, 576, 736, 864, 1024, 1152};

    // 1) pack all layer weights into fragment order (f16)
    for (int i = 0; i < 6; ++i) {
        int total = kpads[i] * 4 * fs[i];
        pack_weights<<<(total + 255) / 256, 256, 0, stream>>>(
            Ws[i], Us[i], wp + wofs[i], cins[i], fs[i], kpads[i]);
    }

    // 2) convert input x f32 -> f16 into bufA
    {
        int nx = 4 * 10 * 128 * 128 * 3;
        f32_to_f16_kernel<<<(nx + 255) / 256, 256, 0, stream>>>(x, bufA, nx);
    }

    // 3) layers (ping-pong bufA/bufB; pools after layers 2,4,6)
    run_layer<128, 128,  3, 32,  320>(bufA, bufB, h0, h1, cst, wp + wofs[0], bs[0], stream);
    run_layer<128, 128, 32, 32,  576>(bufB, bufA, h0, h1, cst, wp + wofs[1], bs[1], stream);
    {   // pool 128 -> 64, C=32 : bufA -> bufB
        int total = 40 * 64 * 64 * 32;
        maxpool2_kernel<<<(total + 255) / 256, 256, 0, stream>>>(
            bufA, bufB, (float*)nullptr, 40, 128, 128, 32);
    }
    run_layer< 64,  64, 32, 48,  736>(bufB, bufA, h0, h1, cst, wp + wofs[2], bs[2], stream);
    run_layer< 64,  64, 48, 48,  864>(bufA, bufB, h0, h1, cst, wp + wofs[3], bs[3], stream);
    {   // pool 64 -> 32, C=48 : bufB -> bufA
        int total = 40 * 32 * 32 * 48;
        maxpool2_kernel<<<(total + 255) / 256, 256, 0, stream>>>(
            bufB, bufA, (float*)nullptr, 40, 64, 64, 48);
    }
    run_layer< 32,  32, 48, 64, 1024>(bufA, bufB, h0, h1, cst, wp + wofs[4], bs[4], stream);
    run_layer< 32,  32, 64, 64, 1152>(bufB, bufA, h0, h1, cst, wp + wofs[5], bs[5], stream);
    {   // final pool 32 -> 16, C=64 : bufA -> d_out (f32)
        int total = 40 * 16 * 16 * 64;   // == out_size
        maxpool2_kernel<<<(total + 255) / 256, 256, 0, stream>>>(
            bufA, (_Float16*)nullptr, (float*)d_out, 40, 32, 32, 64);
    }
}
